// get_model_8744553414808
// MI455X (gfx1250) — matrix operand order repaired and checked by reading, not compiled, not run on hardware
//
#include <hip/hip_runtime.h>

// ---------------------------------------------------------------------------
// CDNA5 / gfx1250 implementation of the spiking point-Mamba reference.
// Heavy GEMMs use v_wmma_f32_16x16x32_bf16 (wave32 WMMA); BN folded into
// weights; spikes are {0,1} so bf16 A-matrices are exact.
// All GEMM dims are tile-aligned (M%128, N%128, K%32) so staging is pure
// b128 vector copies with no EXEC divergence.
// ---------------------------------------------------------------------------

typedef __bf16 bf16;
typedef __attribute__((ext_vector_type(16))) __bf16 v16bf;
typedef __attribute__((ext_vector_type(8)))  float  v8f;
typedef __attribute__((ext_vector_type(4)))  float  fx4;

#define BN_EPS_F 1e-5f

static __host__ __device__ inline size_t cdivz(size_t a, size_t b) { return (a + b - 1) / b; }

// ===========================================================================
// bf16 WMMA GEMM:  C[M,N] = A[M,K] * W[N,K]^T  (+ bias[N])
// Requires: M % 128 == 0, N % 128 == 0, K % 32 == 0 (guaranteed by caller).
// Block = 256 threads = 8 waves; tile 128(M) x 128(N) x 32(K).
// Wave w: M-range (w&3)*32 (2 subtiles), N-half (w>>2)*64 (4 subtiles)
//   -> 8 v_wmma per K-step per wave.
// ===========================================================================
__global__ __launch_bounds__(256)
void k_gemm_bf16(const bf16* __restrict__ A, int lda,
                 const bf16* __restrict__ W, int ldw,
                 float* __restrict__ C, int ldc,
                 const float* __restrict__ bias,
                 int Kn)
{
  __shared__ __align__(16) bf16 lA[128 * 32];   // [m][k]
  __shared__ __align__(16) bf16 lB[128 * 32];   // [n][k]

  const int tid  = threadIdx.x;
  const int lane = tid & 31;
  const int w    = tid >> 5;
  const int wm   = w & 3;       // M 32-row group
  const int wn   = w >> 2;      // N half (64 cols)
  const int h    = lane >> 4;   // lane half
  const int lm   = lane & 15;

  const int m0 = blockIdx.y * 128;
  const int n0 = blockIdx.x * 128;

  // staging: 2 threads per row, 16 contiguous bf16 (32B) each
  const int sr = tid >> 1;
  const int sc = (tid & 1) * 16;
  const bf16* gA = A + (size_t)(m0 + sr) * lda + sc;
  const bf16* gW = W + (size_t)(n0 + sr) * ldw + sc;
  bf16* sA = &lA[sr * 32 + sc];
  bf16* sB = &lB[sr * 32 + sc];

  v8f acc[2][4] = {{{}, {}, {}, {}}, {{}, {}, {}, {}}};

  for (int k0 = 0; k0 < Kn; k0 += 32) {
    // vector staging (global_load_b128 -> ds_store_b128), no bounds checks
    fx4 a0 = ((const fx4*)gA)[0];
    fx4 a1 = ((const fx4*)gA)[1];
    fx4 b0 = ((const fx4*)gW)[0];
    fx4 b1 = ((const fx4*)gW)[1];
    ((fx4*)sA)[0] = a0; ((fx4*)sA)[1] = a1;
    ((fx4*)sB)[0] = b0; ((fx4*)sB)[1] = b1;
    gA += 32; gW += 32;
    if (k0 + 32 < Kn) {             // prefetch next K tile (global_prefetch_b8)
      __builtin_prefetch(gA, 0, 1);
      __builtin_prefetch(gW, 0, 1);
    }
    __syncthreads();

    // A fragment (16-bit A 16x32): lane lm = row; elems e ->
    // K=16h+e (matches B). Two contiguous 16B runs.
    union UF { v16bf v; fx4 f[2]; };
    UF ua[2], ub[4];
#pragma unroll
    for (int a = 0; a < 2; ++a) {
      const bf16* ar = &lA[(wm * 32 + a * 16 + lm) * 32];
      ua[a].f[0] = *(const fx4*)(ar + 16 * h);
      ua[a].f[1] = *(const fx4*)(ar + 16 * h + 8);
    }
    // B fragment: lane lm = column, elems e -> K = 16h + e (32B contiguous)
#pragma unroll
    for (int j = 0; j < 4; ++j) {
      const bf16* br = &lB[(wn * 64 + j * 16 + lm) * 32 + 16 * h];
      ub[j].f[0] = ((const fx4*)br)[0];
      ub[j].f[1] = ((const fx4*)br)[1];
    }
#pragma unroll
    for (int a = 0; a < 2; ++a)
#pragma unroll
      for (int j = 0; j < 4; ++j)
        acc[a][j] = __builtin_amdgcn_wmma_f32_16x16x32_bf16(
            false, ua[a].v, false, ub[j].v, (short)0, acc[a][j], false, false);
    __syncthreads();
  }

  // store: D elem r -> row = r + 8h, col = lm (ISA C/D layout)
#pragma unroll
  for (int j = 0; j < 4; ++j) {
    int n = n0 + wn * 64 + j * 16 + lm;
    float bv = bias ? bias[n] : 0.f;
#pragma unroll
    for (int a = 0; a < 2; ++a) {
#pragma unroll
      for (int r = 0; r < 8; ++r) {
        int gm = m0 + wm * 32 + a * 16 + 8 * h + r;
        C[(size_t)gm * ldc + n] = acc[a][j][r] + bv;
      }
    }
  }
}

// ===========================================================================
// Weight prep: f32 -> bf16, optional per-output-row BN gain fold
// ===========================================================================
__global__ void k_prep_w(const float* __restrict__ W, bf16* __restrict__ dst,
                         const float* __restrict__ gain, int Kd, size_t total)
{
  size_t i = (size_t)blockIdx.x * blockDim.x + threadIdx.x;
  if (i >= total) return;
  int n = (int)(i / Kd);
  float s = gain ? gain[n] * rsqrtf(1.f + BN_EPS_F) : 1.f;
  dst[i] = (bf16)(W[i] * s);
}

__global__ void k_prep_b(const float* __restrict__ be, const float* __restrict__ gain,
                         const float* __restrict__ bb, float* __restrict__ dst, int n)
{
  int i = blockIdx.x * blockDim.x + threadIdx.x;
  if (i >= n) return;
  float s = gain[i] * rsqrtf(1.f + BN_EPS_F);
  dst[i] = be[i] * s + bb[i];
}

// xproj [12][56][768] -> bf16 zero-padded [12][128][768] (rows 56..127 = 0)
__global__ void k_prep_xproj(const float* __restrict__ W, bf16* __restrict__ dst)
{
  size_t i = (size_t)blockIdx.x * blockDim.x + threadIdx.x;
  size_t total = 12ull * 128 * 768;
  if (i >= total) return;
  int k = (int)(i % 768);
  int n = (int)((i / 768) % 128);
  int l = (int)(i / (768 * 128));
  dst[i] = (n < 56) ? (bf16)W[((size_t)l * 56 + n) * 768 + k] : (bf16)0.f;
}

// ===========================================================================
// FPS: one block per batch b (uses t=0 point cloud), serial 64 selections
// ===========================================================================
__global__ __launch_bounds__(256)
void k_fps(const float* __restrict__ xyz, float* __restrict__ centers)
{
  int b = blockIdx.x;
  __shared__ float px[1024], py[1024], pz[1024], dist[1024];
  __shared__ float bv[256];
  __shared__ int   bi[256];
  __shared__ int   s_last;
  const float* P = xyz + (size_t)b * 1024 * 3;   // t=0 slice
  for (int i = threadIdx.x; i < 1024; i += 256) {
    px[i] = P[i * 3 + 0]; py[i] = P[i * 3 + 1]; pz[i] = P[i * 3 + 2];
    dist[i] = 1e10f;
  }
  if (threadIdx.x == 0) s_last = 0;
  __syncthreads();
  for (int g = 0; g < 64; ++g) {
    int last = s_last;
    float cx = px[last], cy = py[last], cz = pz[last];
    if (threadIdx.x == 0) {
      float* c = centers + ((size_t)b * 64 + g) * 3;
      c[0] = cx; c[1] = cy; c[2] = cz;
    }
    float mv = -1.f; int mi = 0;
    for (int i = threadIdx.x; i < 1024; i += 256) {
      float dx = px[i] - cx, dy = py[i] - cy, dz = pz[i] - cz;
      float nd = fminf(dist[i], dx * dx + dy * dy + dz * dz);
      dist[i] = nd;
      if (nd > mv || (nd == mv && i < mi)) { mv = nd; mi = i; }
    }
    bv[threadIdx.x] = mv; bi[threadIdx.x] = mi;
    __syncthreads();
    for (int s = 128; s > 0; s >>= 1) {
      if (threadIdx.x < s) {
        float ov = bv[threadIdx.x + s]; int oi = bi[threadIdx.x + s];
        if (ov > bv[threadIdx.x] || (ov == bv[threadIdx.x] && oi < bi[threadIdx.x])) {
          bv[threadIdx.x] = ov; bi[threadIdx.x] = oi;
        }
      }
      __syncthreads();
    }
    if (threadIdx.x == 0) s_last = bi[0];
    __syncthreads();
  }
}

// ===========================================================================
// KNN: block per (g, tb); pick 32 nearest, write nb = p[idx] - center
// ===========================================================================
__global__ __launch_bounds__(256)
void k_knn(const float* __restrict__ xyz, const float* __restrict__ centers,
           float* __restrict__ nb)
{
  int g  = blockIdx.x;
  int tb = blockIdx.y;
  int b  = tb & 15;
  __shared__ float px[1024], py[1024], pz[1024], dist[1024];
  __shared__ float bv[256];
  __shared__ int   bi[256];
  const float* P = xyz + (size_t)tb * 1024 * 3;
  const float* c = centers + ((size_t)b * 64 + g) * 3;
  float cx = c[0], cy = c[1], cz = c[2];
  for (int i = threadIdx.x; i < 1024; i += 256) {
    float x = P[i * 3 + 0], y = P[i * 3 + 1], z = P[i * 3 + 2];
    px[i] = x; py[i] = y; pz[i] = z;
    float dx = x - cx, dy = y - cy, dz = z - cz;
    dist[i] = dx * dx + dy * dy + dz * dz;
  }
  __syncthreads();
  for (int m = 0; m < 32; ++m) {
    float mv = 1e30f; int mi = 0;
    for (int i = threadIdx.x; i < 1024; i += 256) {
      float d = dist[i];
      if (d < mv || (d == mv && i < mi)) { mv = d; mi = i; }
    }
    bv[threadIdx.x] = mv; bi[threadIdx.x] = mi;
    __syncthreads();
    for (int s = 128; s > 0; s >>= 1) {
      if (threadIdx.x < s) {
        float ov = bv[threadIdx.x + s]; int oi = bi[threadIdx.x + s];
        if (ov < bv[threadIdx.x] || (ov == bv[threadIdx.x] && oi < bi[threadIdx.x])) {
          bv[threadIdx.x] = ov; bi[threadIdx.x] = oi;
        }
      }
      __syncthreads();
    }
    if (threadIdx.x == 0) {
      int s = bi[0];
      dist[s] = 1e30f;
      float* o = nb + (((size_t)tb * 64 + g) * 32 + m) * 3;
      o[0] = px[s] - cx; o[1] = py[s] - cy; o[2] = pz[s] - cz;
    }
    __syncthreads();
  }
}

// ===========================================================================
// enc1 (K=3) scalar GEMM with BN folded: out[65536][128]
// ===========================================================================
__global__ void k_enc1(const float* __restrict__ nb, const float* __restrict__ w,
                       const float* __restrict__ be, const float* __restrict__ g,
                       const float* __restrict__ bb, float* __restrict__ out)
{
  size_t i = (size_t)blockIdx.x * blockDim.x + threadIdx.x;
  if (i >= 65536ull * 128) return;
  int c = (int)(i & 127);
  size_t r = i >> 7;
  const float* p = nb + r * 3;
  float s = g[c] * rsqrtf(1.f + BN_EPS_F);
  float acc = p[0] * w[c * 3] + p[1] * w[c * 3 + 1] + p[2] * w[c * 3 + 2] + be[c];
  out[i] = acc * s + bb[c];
}

// ===========================================================================
// LIF over T=2 (tau=2, v_th=0.5, hard reset): pairs rows (r, r+pairOff)
// ===========================================================================
__global__ void k_lif(const float* __restrict__ pre, int ldPre,
                      bf16* __restrict__ out, int ldOut,
                      int cols, int pairOff, size_t total)
{
  size_t i = (size_t)blockIdx.x * blockDim.x + threadIdx.x;
  if (i >= total) return;
  int c = (int)(i % cols);
  size_t r  = i / cols;
  size_t r2 = r + (size_t)pairOff;
  float x0 = pre[r  * (size_t)ldPre + c];
  float x1 = pre[r2 * (size_t)ldPre + c];
  float v  = 0.5f * x0;
  float s0 = (v >= 0.5f) ? 1.f : 0.f;
  v *= (1.f - s0);
  v = 0.5f * (v + x1);
  float s1 = (v >= 0.5f) ? 1.f : 0.f;
  out[r  * (size_t)ldOut + c] = (bf16)s0;
  out[r2 * (size_t)ldOut + c] = (bf16)s1;
}

// broadcast global-feature spikes into concat cols 0..255 of s2cat (ld 512)
__global__ void k_bcast_gl(const bf16* __restrict__ glspk, bf16* __restrict__ s2cat)
{
  size_t i = (size_t)blockIdx.x * blockDim.x + threadIdx.x;
  if (i >= 65536ull * 256) return;
  int c = (int)(i & 255);
  size_t r = i >> 8;
  s2cat[r * 512 + c] = glspk[(r >> 5) * 256 + c];
}

// max over M=32 neighbor rows
__global__ void k_maxM(const float* __restrict__ pre, int cols,
                       float* __restrict__ out, size_t total)
{
  size_t i = (size_t)blockIdx.x * blockDim.x + threadIdx.x;
  if (i >= total) return;
  int c = (int)(i % cols);
  size_t R = i / cols;
  const float* p = pre + R * 32 * (size_t)cols + c;
  float m = p[0];
#pragma unroll
  for (int j = 1; j < 32; ++j) m = fmaxf(m, p[(size_t)j * cols]);
  out[R * cols + c] = m;
}

// pos embedding: hidden[row] += pos2(gelu(pos1(center)))
__global__ __launch_bounds__(128)
void k_pos(const float* __restrict__ centers,
           const float* __restrict__ w1, const float* __restrict__ b1,
           const float* __restrict__ w2, const float* __restrict__ b2,
           float* __restrict__ hidden)
{
  int row = blockIdx.x;                 // 0..2047
  int tb = row >> 6, g = row & 63, b = tb & 15;
  const float* c = centers + ((size_t)b * 64 + g) * 3;
  float cx = c[0], cy = c[1], cz = c[2];
  __shared__ float h1[128];
  int j = threadIdx.x;
  float x = cx * w1[j * 3] + cy * w1[j * 3 + 1] + cz * w1[j * 3 + 2] + b1[j];
  float t = tanhf(0.7978845608028654f * (x + 0.044715f * x * x * x));
  h1[j] = 0.5f * x * (1.f + t);
  __syncthreads();
  for (int c2 = j; c2 < 384; c2 += 128) {
    float acc = b2[c2];
    const float* wr = w2 + (size_t)c2 * 128;
#pragma unroll 8
    for (int k = 0; k < 128; ++k) acc += h1[k] * wr[k];
    hidden[(size_t)row * 384 + c2] += acc;
  }
}

// residual = hidden (first) or hidden + residual
__global__ void k_resid(const float* __restrict__ hidden, float* __restrict__ residual,
                        int first, size_t total)
{
  size_t i = (size_t)blockIdx.x * blockDim.x + threadIdx.x;
  if (i >= total) return;
  residual[i] = first ? hidden[i] : hidden[i] + residual[i];
}

// depthwise causal conv (D_CONV=4) + LIF over T -> spikes xs [2048][768]
__global__ void k_convlif(const float* __restrict__ xz,        // [2048][1536]; xh cols 0..767
                          const float* __restrict__ cw,        // [768][4]
                          const float* __restrict__ cb,
                          bf16* __restrict__ xs)
{
  size_t i = (size_t)blockIdx.x * blockDim.x + threadIdx.x;
  if (i >= 16ull * 64 * 768) return;
  int d = (int)(i % 768);
  size_t rem = i / 768;
  int l = (int)(rem % 64);
  int b = (int)(rem / 64);
  float w0 = cw[d * 4], w1 = cw[d * 4 + 1], w2 = cw[d * 4 + 2], w3 = cw[d * 4 + 3];
  float bias = cb[d];
  float cv[2];
#pragma unroll
  for (int t = 0; t < 2; ++t) {
    int tb = t * 16 + b;
    const float* X = xz + (size_t)tb * 64 * 1536 + d;
    float a = bias;
    if (l >= 3) a += w0 * X[(size_t)(l - 3) * 1536];
    if (l >= 2) a += w1 * X[(size_t)(l - 2) * 1536];
    if (l >= 1) a += w2 * X[(size_t)(l - 1) * 1536];
    a += w3 * X[(size_t)l * 1536];
    cv[t] = a;
  }
  float v  = 0.5f * cv[0];
  float s0 = (v >= 0.5f) ? 1.f : 0.f;
  v *= (1.f - s0);
  v = 0.5f * (v + cv[1]);
  float s1 = (v >= 0.5f) ? 1.f : 0.f;
  xs[((size_t)b        * 64 + l) * 768 + d] = (bf16)s0;
  xs[((size_t)(16 + b) * 64 + l) * 768 + d] = (bf16)s1;
}

// dt = softplus(dbl[:, :24] @ dt_w^T + dt_b)   (K=24 scalar, precision-clean)
__global__ void k_dt(const float* __restrict__ dbl, const float* __restrict__ dw,
                     const float* __restrict__ db, float* __restrict__ dt)
{
  size_t i = (size_t)blockIdx.x * blockDim.x + threadIdx.x;
  if (i >= 2048ull * 768) return;
  int d = (int)(i % 768);
  size_t r = i / 768;
  const float* x = dbl + r * 128;
  const float* w = dw + (size_t)d * 24;
  float a = db[d];
#pragma unroll
  for (int k = 0; k < 24; ++k) a += x[k] * w[k];
  dt[i] = fmaxf(a, 0.f) + log1pf(expf(-fabsf(a)));
}

// selective scan: thread per (tb, d); 16 states over L=64; fused *lif(z), Dp
__global__ __launch_bounds__(256)
void k_scan(const float* __restrict__ dt, const bf16* __restrict__ xs,
            const float* __restrict__ dbl, const float* __restrict__ A_log,
            const float* __restrict__ Dp, const bf16* __restrict__ zs,
            bf16* __restrict__ yb)
{
  int d  = blockIdx.x * 256 + threadIdx.x;    // 0..767
  int tb = blockIdx.y;                        // 0..31
  float Av[16];
#pragma unroll
  for (int s = 0; s < 16; ++s) Av[s] = -expf(A_log[(size_t)d * 16 + s]);
  float Dpd = Dp[d];
  float h[16] = {};
  for (int l = 0; l < 64; ++l) {
    size_t row = (size_t)tb * 64 + l;
    float dtv = dt[row * 768 + d];
    float xv  = (float)xs[row * 768 + d];
    float dx  = dtv * xv;
    const float* bc = dbl + row * 128;
    float y = 0.f;
#pragma unroll
    for (int s = 0; s < 16; ++s) {
      h[s] = h[s] * expf(dtv * Av[s]) + dx * bc[24 + s];
      y += h[s] * bc[40 + s];
    }
    y += xv * Dpd;
    y *= (float)zs[row * 768 + d];
    yb[row * 768 + d] = (bf16)y;
  }
}

// final: layernorm(hidden + residual) -> d_out
__global__ __launch_bounds__(128)
void k_final(const float* __restrict__ hidden, const float* __restrict__ residual,
             const float* __restrict__ g, const float* __restrict__ b,
             float* __restrict__ out)
{
  int row = blockIdx.x;
  __shared__ float buf[384];
  __shared__ float red[128];
  int t = threadIdx.x;
  float s = 0.f;
  for (int c = t; c < 384; c += 128) {
    float x = hidden[(size_t)row * 384 + c] + residual[(size_t)row * 384 + c];
    buf[c] = x; s += x;
  }
  red[t] = s; __syncthreads();
  for (int st = 64; st > 0; st >>= 1) { if (t < st) red[t] += red[t + st]; __syncthreads(); }
  float mu = red[0] / 384.f;
  __syncthreads();
  float s2 = 0.f;
  for (int c = t; c < 384; c += 128) { float dx = buf[c] - mu; s2 += dx * dx; }
  red[t] = s2; __syncthreads();
  for (int st = 64; st > 0; st >>= 1) { if (t < st) red[t] += red[t + st]; __syncthreads(); }
  float inv = rsqrtf(red[0] / 384.f + 1e-5f);
  for (int c = t; c < 384; c += 128)
    out[(size_t)row * 384 + c] = (buf[c] - mu) * inv * g[c] + b[c];
}

// ===========================================================================
// Host orchestration
// ===========================================================================
static void gemm(const bf16* A, int lda, const bf16* W, int ldw, float* C, int ldc,
                 const float* bias, int Mn, int Nn, int Kn, hipStream_t s)
{
  // caller guarantees Mn%128==0, Nn%128==0, Kn%32==0
  dim3 grid((unsigned)(Nn / 128), (unsigned)(Mn / 128));
  k_gemm_bf16<<<grid, 256, 0, s>>>(A, lda, W, ldw, C, ldc, bias, Kn);
}

#define EW(total) dim3((unsigned)cdivz((size_t)(total), 256)), dim3(256), 0, stream

extern "C" void kernel_launch(void* const* d_in, const int* in_sizes, int n_in,
                              void* d_out, int out_size, void* d_ws, size_t ws_size,
                              hipStream_t stream)
{
  (void)in_sizes; (void)n_in; (void)out_size; (void)ws_size;
  const float* xyz     = (const float*)d_in[0];
  const float* enc1_w  = (const float*)d_in[1];
  const float* enc1_b  = (const float*)d_in[2];
  const float* bn1_g   = (const float*)d_in[3];
  const float* bn1_b   = (const float*)d_in[4];
  const float* enc2_w  = (const float*)d_in[5];
  const float* enc2_b  = (const float*)d_in[6];
  const float* bn2_g   = (const float*)d_in[7];
  const float* bn2_b   = (const float*)d_in[8];
  const float* enc3_w  = (const float*)d_in[9];
  const float* enc3_b  = (const float*)d_in[10];
  const float* bn3_g   = (const float*)d_in[11];
  const float* bn3_b   = (const float*)d_in[12];
  const float* enc4_w  = (const float*)d_in[13];
  const float* enc4_b  = (const float*)d_in[14];
  const float* bn4_g   = (const float*)d_in[15];
  const float* bn4_b   = (const float*)d_in[16];
  const float* pos1_w  = (const float*)d_in[17];
  const float* pos1_b  = (const float*)d_in[18];
  const float* pos2_w  = (const float*)d_in[19];
  const float* pos2_b  = (const float*)d_in[20];
  const float* in_w    = (const float*)d_in[21];
  const float* conv_w  = (const float*)d_in[22];
  const float* conv_b  = (const float*)d_in[23];
  const float* xproj_w = (const float*)d_in[24];
  const float* dt_w    = (const float*)d_in[25];
  const float* dt_b    = (const float*)d_in[26];
  const float* A_log   = (const float*)d_in[27];
  const float* Dp      = (const float*)d_in[28];
  const float* out_w   = (const float*)d_in[29];
  const float* normf_g = (const float*)d_in[30];
  const float* normf_b = (const float*)d_in[31];
  float* outp = (float*)d_out;

  // ---- workspace arena ----
  char* base = (char*)d_ws;
  size_t off = 0;
  auto alloc = [&](size_t bytes) -> void* {
    void* r = base + off;
    off += (bytes + 255) & ~(size_t)255;
    return r;
  };
  bf16*  wE2   = (bf16*)alloc(256ull * 128 * 2);
  bf16*  wE3   = (bf16*)alloc(512ull * 512 * 2);
  bf16*  wE4   = (bf16*)alloc(384ull * 512 * 2);
  bf16*  wIn   = (bf16*)alloc(12ull * 1536 * 384 * 2);
  bf16*  wXp   = (bf16*)alloc(12ull * 128 * 768 * 2);   // N padded 56 -> 128
  bf16*  wOut  = (bf16*)alloc(12ull * 384 * 768 * 2);
  float* bE2   = (float*)alloc(256 * 4);
  float* bE3   = (float*)alloc(512 * 4);
  float* bE4   = (float*)alloc(384 * 4);
  float* centers = (float*)alloc(16ull * 64 * 3 * 4);
  float* nb      = (float*)alloc(65536ull * 3 * 4);
  float* gl      = (float*)alloc(2048ull * 256 * 4);
  bf16*  glspk   = (bf16*)alloc(2048ull * 256 * 2);
  float* hidden  = (float*)alloc(2048ull * 384 * 4);
  float* residual= (float*)alloc(2048ull * 384 * 4);
  float* preA    = (float*)alloc(65536ull * 512 * 4);   // 128 MB pre-act arena
  bf16*  spkB    = (bf16*)alloc(65536ull * 512 * 2);    //  64 MB spike arena
  // Mamba buffers carved from preA (encoder done by then)
  float* xzF  = preA;                         // 2048*1536 f32
  float* dblF = preA + 3200000;               // 2048*128  f32
  float* dtF  = preA + 3500000;               // 2048*768  f32
  bf16*  xsB  = (bf16*)(preA + 5100000);      // 2048*768  bf16
  bf16*  zsB  = (bf16*)(preA + 5900000);
  bf16*  ybB  = (bf16*)(preA + 6700000);
  bf16*  hsB  = (bf16*)(preA + 7500000);      // 2048*384  bf16

  // ---- weight prep (BN folded into enc weights/biases) ----
  k_prep_w<<<EW(256ull*128)>>>(enc2_w, wE2, bn2_g, 128, 256ull * 128);
  k_prep_w<<<EW(512ull*512)>>>(enc3_w, wE3, bn3_g, 512, 512ull * 512);
  k_prep_w<<<EW(384ull*512)>>>(enc4_w, wE4, bn4_g, 512, 384ull * 512);
  k_prep_w<<<EW(12ull*1536*384)>>>(in_w,  wIn,  nullptr, 384, 12ull * 1536 * 384);
  k_prep_w<<<EW(12ull*384*768)>>>(out_w, wOut, nullptr, 768, 12ull * 384 * 768);
  k_prep_xproj<<<EW(12ull*128*768)>>>(xproj_w, wXp);
  k_prep_b<<<1, 256, 0, stream>>>(enc2_b, bn2_g, bn2_b, bE2, 256);
  k_prep_b<<<2, 256, 0, stream>>>(enc3_b, bn3_g, bn3_b, bE3, 512);
  k_prep_b<<<2, 256, 0, stream>>>(enc4_b, bn4_g, bn4_b, bE4, 384);

  // ---- grouping: FPS + KNN ----
  k_fps<<<16, 256, 0, stream>>>(xyz, centers);
  k_knn<<<dim3(64, 32), 256, 0, stream>>>(xyz, centers, nb);

  // ---- encoder ----
  k_enc1<<<EW(65536ull*128)>>>(nb, enc1_w, enc1_b, bn1_g, bn1_b, preA);
  k_lif<<<EW(32768ull*128)>>>(preA, 128, spkB, 128, 128, 32768, 32768ull * 128);
  gemm(spkB, 128, wE2, 128, preA, 256, bE2, 65536, 256, 128, stream);
  k_maxM<<<EW(2048ull*256)>>>(preA, 256, gl, 2048ull * 256);
  k_lif<<<EW(1024ull*256)>>>(gl, 256, glspk, 256, 256, 1024, 1024ull * 256);
  k_lif<<<EW(32768ull*256)>>>(preA, 256, spkB + 256, 512, 256, 32768, 32768ull * 256);
  k_bcast_gl<<<EW(65536ull*256)>>>(glspk, spkB);
  gemm(spkB, 512, wE3, 512, preA, 512, bE3, 65536, 512, 512, stream);
  k_lif<<<EW(32768ull*512)>>>(preA, 512, spkB, 512, 512, 32768, 32768ull * 512);
  gemm(spkB, 512, wE4, 512, preA, 384, bE4, 65536, 384, 512, stream);
  k_maxM<<<EW(2048ull*384)>>>(preA, 384, hidden, 2048ull * 384);
  k_pos<<<2048, 128, 0, stream>>>(centers, pos1_w, pos1_b, pos2_w, pos2_b, hidden);

  // ---- 12 spiking-Mamba layers ----
  for (int l = 0; l < 12; ++l) {
    k_resid<<<EW(2048ull*384)>>>(hidden, residual, l == 0 ? 1 : 0, 2048ull * 384);
    k_lif<<<EW(1024ull*384)>>>(residual, 384, hsB, 384, 384, 1024, 1024ull * 384);
    gemm(hsB, 384, wIn + (size_t)l * 1536 * 384, 384, xzF, 1536, nullptr, 2048, 1536, 384, stream);
    k_convlif<<<EW(16ull*64*768)>>>(xzF, conv_w + (size_t)l * 768 * 4, conv_b + (size_t)l * 768, xsB);
    gemm(xsB, 768, wXp + (size_t)l * 128 * 768, 768, dblF, 128, nullptr, 2048, 128, 768, stream);
    k_dt<<<EW(2048ull*768)>>>(dblF, dt_w + (size_t)l * 768 * 24, dt_b + (size_t)l * 768, dtF);
    k_lif<<<EW(1024ull*768)>>>(xzF + 768, 1536, zsB, 768, 768, 1024, 1024ull * 768);
    k_scan<<<dim3(3, 32), 256, 0, stream>>>(dtF, xsB, dblF, A_log + (size_t)l * 768 * 16,
                                            Dp + (size_t)l * 768, zsB, ybB);
    gemm(ybB, 768, wOut + (size_t)l * 384 * 768, 768, hidden, 384, nullptr, 2048, 384, 768, stream);
  }

  // ---- final layernorm(hidden + residual) ----
  k_final<<<2048, 128, 0, stream>>>(hidden, residual, normf_g, normf_b, outp);
}
